// MultiHeadedAttention_23931557773360
// MI455X (gfx1250) — matrix operand order repaired and checked by reading, not compiled, not run on hardware
//
#include <hip/hip_runtime.h>

// Problem constants
#define B_  8
#define S_  1024
#define D_  1024
#define H_  8
#define DK_ 128
#define RNK 16
#define HD_ 1024   // H*DK

typedef __attribute__((ext_vector_type(16))) __bf16          v16bf;
typedef __attribute__((ext_vector_type(8)))  float           v8f;
typedef __attribute__((ext_vector_type(8)))  unsigned short  us8;
typedef __attribute__((ext_vector_type(4)))  unsigned short  us4;
typedef __attribute__((ext_vector_type(4)))  float           f4;
typedef __attribute__((ext_vector_type(4)))  int             v4i;

union FragBF {          // one WMMA bf16 A/B operand: 16 bf16 per lane = 8 VGPRs
  v16bf v;
  us8   h[2];
};

// ---- CDNA5 async global->LDS copy (ASYNCcnt), guarded with sync fallback ----
#if __has_builtin(__builtin_amdgcn_global_load_async_to_lds_b128) && \
    __has_builtin(__builtin_amdgcn_s_wait_asynccnt)
#define ASYNC_LDS 1
#else
#define ASYNC_LDS 0
#endif

__device__ __forceinline__ void cp16_lds(void* l, const void* g) {
#if ASYNC_LDS
  __builtin_amdgcn_global_load_async_to_lds_b128(
      (__attribute__((address_space(1))) v4i*)g,
      (__attribute__((address_space(3))) v4i*)l, 0, 0);
#else
  *(us8*)l = *(const us8*)g;
#endif
}
__device__ __forceinline__ void wait_async_lds() {
#if ASYNC_LDS
  __builtin_amdgcn_s_wait_asynccnt(0);
#endif
}

__device__ __forceinline__ unsigned short f2bf(float x) {   // native RNE cvt (1 VALU op)
  union { __bf16 b; unsigned short u; } c;
  c.b = (__bf16)x;
  return c.u;
}

__device__ __forceinline__ float redmax16(float x) {
  #pragma unroll
  for (int m = 1; m < 16; m <<= 1) x = fmaxf(x, __shfl_xor(x, m, 16));
  return x;
}
__device__ __forceinline__ float redsum16(float x) {
  #pragma unroll
  for (int m = 1; m < 16; m <<= 1) x += __shfl_xor(x, m, 16);
  return x;
}

// ---------------------------------------------------------------------------
// f32 -> bf16 bulk convert (4 elements / thread)
// ---------------------------------------------------------------------------
__global__ __launch_bounds__(256) void f32_to_bf16_kernel(
    const float* __restrict__ in, unsigned short* __restrict__ out) {
  long i = ((long)blockIdx.x * blockDim.x + threadIdx.x) * 4;
  f4 x = *(const f4*)&in[i];
  us4 y;
  #pragma unroll
  for (int k = 0; k < 4; ++k) y[k] = f2bf(x[k]);
  *(us4*)&out[i] = y;
}

// ---------------------------------------------------------------------------
// W' = Wbase + WA @ WB  (rank-RNK LoRA fold), emitted as bf16. rank==0 => convert.
// ---------------------------------------------------------------------------
__global__ __launch_bounds__(256) void fuse_weights_kernel(
    const float* __restrict__ Wbase, const float* __restrict__ WA,
    const float* __restrict__ WB, unsigned short* __restrict__ out,
    int rank, int n) {
  int idx = blockIdx.x * blockDim.x + threadIdx.x;
  int d = idx / n, c = idx % n;
  float acc = Wbase[idx];
  for (int r = 0; r < rank; ++r)
    acc += WA[d * rank + r] * WB[r * n + c];
  out[idx] = f2bf(acc);
}

// ---------------------------------------------------------------------------
// C[M,N] = A[M,K] @ W[K,N] + bias ; A,W bf16; double-buffered async LDS staging.
// OUT_MODE: 0 = f32 row-major, 1 = bf16 row-major, 2 = bf16 [b][h][d][s] (K^T).
// 128x128 block tile, 8 waves, each wave 32x64 (2x4 WMMA tiles).
// ---------------------------------------------------------------------------
template<int OUT_MODE>
__global__ __launch_bounds__(256, 1) void gemm_bias_kernel(
    const unsigned short* __restrict__ A, const unsigned short* __restrict__ W,
    const float* __restrict__ bias, void* __restrict__ Outp, int N, int K) {
  __shared__ unsigned short As[2][128 * 32];
  __shared__ unsigned short Ws[2][32 * 128];
  const int ntile = N >> 7;
  const int bm = blockIdx.x / ntile;
  const int bn = blockIdx.x % ntile;
  const int tid  = threadIdx.x;
  const int lane = tid & 31, wave = tid >> 5;
  const int hh = lane >> 4, l16 = lane & 15;
  const int wm = wave & 3;    // 4 row groups of 32
  const int wn = wave >> 2;   // 2 col groups of 64

  const v8f zero = {};
  v8f acc[2][4];
  #pragma unroll
  for (int i = 0; i < 2; ++i)
    #pragma unroll
    for (int j = 0; j < 4; ++j) acc[i][j] = zero;

  auto stage = [&](int buf, int kk) {
    #pragma unroll
    for (int i = 0; i < 2; ++i) {           // A tile 128x32: 512 16B chunks
      int c = tid + i * 256;
      int r = c >> 2, off = (c & 3) * 8;
      cp16_lds(&As[buf][r * 32 + off], &A[(long)(bm * 128 + r) * K + kk + off]);
    }
    #pragma unroll
    for (int i = 0; i < 2; ++i) {           // W tile 32x128: 512 16B chunks
      int c = tid + i * 256;
      int r = c >> 4, off = (c & 15) * 8;
      cp16_lds(&Ws[buf][r * 128 + off], &W[(long)(kk + r) * N + bn * 128 + off]);
    }
  };

  stage(0, 0);
  for (int kk = 0; kk < K; kk += 32) {
    int buf = (kk >> 5) & 1;
    wait_async_lds();
    __syncthreads();                        // staged tile visible to all waves
    if (kk + 32 < K) stage(buf ^ 1, kk + 32);

    FragBF a[2], bfr[4];
    #pragma unroll
    for (int i = 0; i < 2; ++i) {           // A frag: row = l16, K split per ISA layout
      int row = wm * 32 + i * 16 + l16;
      a[i].h[0] = *(const us8*)&As[buf][row * 32 + hh * 8];
      a[i].h[1] = *(const us8*)&As[buf][row * 32 + 16 + hh * 8];
    }
    #pragma unroll
    for (int j = 0; j < 4; ++j) {           // B frag: col = l16, K split per ISA layout
      int col = wn * 64 + j * 16 + l16;
      #pragma unroll
      for (int s = 0; s < 16; ++s)
        bfr[j].h[s >> 3][s & 7] = Ws[buf][((s & 7) + hh * 8 + (s >> 3) * 16) * 128 + col];
    }
    #pragma unroll
    for (int i = 0; i < 2; ++i)
      #pragma unroll
      for (int j = 0; j < 4; ++j)
        acc[i][j] = __builtin_amdgcn_wmma_f32_16x16x32_bf16(
            false, a[i].v, false, bfr[j].v, (short)0, acc[i][j], false, false);
  }

  #pragma unroll
  for (int i = 0; i < 2; ++i)
    #pragma unroll
    for (int j = 0; j < 4; ++j) {
      int colg = bn * 128 + wn * 64 + j * 16 + l16;
      float bval = bias ? bias[colg] : 0.0f;
      #pragma unroll
      for (int v = 0; v < 8; ++v) {
        int rowg = bm * 128 + wm * 32 + i * 16 + v + 8 * hh;
        float val = acc[i][j][v] + bval;
        if constexpr (OUT_MODE == 0) {
          ((float*)Outp)[(long)rowg * N + colg] = val;
        } else if constexpr (OUT_MODE == 1) {
          ((unsigned short*)Outp)[(long)rowg * N + colg] = f2bf(val);
        } else {                            // K^T layout [b][h][d][s]
          int bb = rowg >> 10, s = rowg & (S_ - 1);
          int hc = colg >> 7,  dl = colg & (DK_ - 1);
          ((unsigned short*)Outp)[((long)(bb * H_ + hc) * DK_ + dl) * S_ + s] = f2bf(val);
        }
      }
    }
}

// ---------------------------------------------------------------------------
// Flash attention: block = (b, h, 128-query tile); wave owns 16 query rows.
// K comes pre-transposed ([b][h][d][s]); mask tile staged to LDS per K-block.
// ---------------------------------------------------------------------------
__global__ __launch_bounds__(256, 1) void attention_kernel(
    const unsigned short* __restrict__ Qg,
    const unsigned short* __restrict__ Kt,   // [b][h][d][s]
    const unsigned short* __restrict__ Vg,   // [b][s][h*DK]
    const unsigned char* __restrict__ mask,
    unsigned short* __restrict__ Xg) {
  __shared__ unsigned short sKT[128 * 64];    // K^T tile: [d][key]        16 KB
  __shared__ unsigned short sV [64 * 128];    // V tile:   [key][d]        16 KB
  __shared__ unsigned short sP [8 * 16 * 32]; // per-wave P staging         8 KB
  __shared__ unsigned char  sM [128 * 64];    // mask tile [qrow][key]      8 KB

  const int nQB = S_ / 128;
  const int b  = blockIdx.x / (H_ * nQB);
  const int h  = (blockIdx.x / nQB) % H_;
  const int qb = (blockIdx.x % nQB) * 128;
  const int tid  = threadIdx.x;
  const int lane = tid & 31, wave = tid >> 5;
  const int hh = lane >> 4, l16 = lane & 15;
  const int qrow0 = qb + wave * 16;

  // Q fragments resident in registers for all DK=128 (4 K-steps of 32)
  FragBF qf[4];
  {
    const unsigned short* qbase = Qg + (long)(b * S_ + qrow0 + l16) * HD_ + h * DK_;
    #pragma unroll
    for (int f = 0; f < 4; ++f) {
      qf[f].h[0] = *(const us8*)&qbase[f * 32 + hh * 8];
      qf[f].h[1] = *(const us8*)&qbase[f * 32 + 16 + hh * 8];
    }
  }

  const v8f zero = {};
  v8f o[8];
  float m[8], l[8];
  #pragma unroll
  for (int t = 0; t < 8; ++t) { o[t] = zero; m[t] = -3.0e38f; l[t] = 0.0f; }
  const float NEGF = -2147483648.0f;

  const unsigned short* kbase = Kt + (long)(b * H_ + h) * DK_ * S_;
  const unsigned char*  mbase = mask + ((long)b * S_ + qb) * S_;

  for (int kb = 0; kb < S_; kb += 64) {
    #pragma unroll
    for (int i = 0; i < 4; ++i) {           // stage K^T tile: 1024 16B chunks
      int c = tid + i * 256;
      int d = c >> 3, koff = (c & 7) * 8;
      cp16_lds(&sKT[d * 64 + koff], &kbase[(long)d * S_ + kb + koff]);
    }
    #pragma unroll
    for (int i = 0; i < 4; ++i) {           // stage V tile: 1024 16B chunks
      int c = tid + i * 256;
      int key = c >> 4, doff = (c & 15) * 8;
      cp16_lds(&sV[key * 128 + doff],
               &Vg[(long)(b * S_ + kb + key) * HD_ + h * DK_ + doff]);
    }
    #pragma unroll
    for (int i = 0; i < 2; ++i) {           // stage mask tile: 512 16B chunks
      int c = tid + i * 256;
      int r = c >> 2, koff = (c & 3) * 16;
      cp16_lds(&sM[r * 64 + koff], &mbase[(long)r * S_ + kb + koff]);
    }
    wait_async_lds();
    __syncthreads();

    #pragma unroll
    for (int j = 0; j < 64; j += 32) {      // 32 keys per inner step
      v8f s0 = zero, s1 = zero;             // scores: keys j..j+15 / j+16..j+31
      #pragma unroll
      for (int f = 0; f < 4; ++f) {
        FragBF b0, b1;
        const unsigned short* kt = &sKT[(f * 32) * 64 + j + l16];
        #pragma unroll
        for (int s = 0; s < 16; ++s) {
          int kd = (s & 7) + hh * 8 + (s >> 3) * 16;
          b0.h[s >> 3][s & 7] = kt[kd * 64];
          b1.h[s >> 3][s & 7] = kt[kd * 64 + 16];
        }
        s0 = __builtin_amdgcn_wmma_f32_16x16x32_bf16(false, qf[f].v, false, b0.v, (short)0, s0, false, false);
        s1 = __builtin_amdgcn_wmma_f32_16x16x32_bf16(false, qf[f].v, false, b1.v, (short)0, s1, false, false);
      }
      // masked online softmax per owned row (C layout: row = v + 8*hh, col = l16)
      #pragma unroll
      for (int v = 0; v < 8; ++v) {
        const unsigned char* mrow = &sM[(wave * 16 + v + 8 * hh) * 64 + j];
        float a0 = mrow[l16]      ? s0[v] : NEGF;
        float a1 = mrow[16 + l16] ? s1[v] : NEGF;
        float mn = fmaxf(m[v], redmax16(fmaxf(a0, a1)));
        float corr = __expf(m[v] - mn);
        m[v] = mn;
        float p0 = __expf(a0 - mn);
        float p1 = __expf(a1 - mn);
        l[v] = l[v] * corr + redsum16(p0 + p1);
        #pragma unroll
        for (int t = 0; t < 8; ++t) o[t][v] *= corr;
        unsigned short* prow = &sP[(wave * 16 + v + 8 * hh) * 32];
        prow[l16]      = f2bf(p0);
        prow[16 + l16] = f2bf(p1);
      }
      // rebuild P as an A-fragment (wave-private LDS region; LDS in-order per wave)
      FragBF pf;
      {
        const unsigned short* pr = &sP[(wave * 16 + l16) * 32];
        pf.h[0] = *(const us8*)&pr[hh * 8];
        pf.h[1] = *(const us8*)&pr[16 + hh * 8];
      }
      #pragma unroll
      for (int t = 0; t < 8; ++t) {         // O(16x128) += P(16x32) @ V(32x128)
        FragBF bv;
        const unsigned short* vr = &sV[j * 128 + t * 16 + l16];
        #pragma unroll
        for (int s = 0; s < 16; ++s)
          bv.h[s >> 3][s & 7] = vr[((s & 7) + hh * 8 + (s >> 3) * 16) * 128];
        o[t] = __builtin_amdgcn_wmma_f32_16x16x32_bf16(false, pf.v, false, bv.v, (short)0, o[t], false, false);
      }
    }
    __syncthreads();                        // protect sKT/sV/sM before next stage
  }

  #pragma unroll
  for (int v = 0; v < 8; ++v) {
    float inv = 1.0f / l[v];
    int rowg = b * S_ + qrow0 + v + 8 * hh;
    #pragma unroll
    for (int t = 0; t < 8; ++t)
      Xg[(long)rowg * HD_ + h * DK_ + t * 16 + l16] = f2bf(o[t][v] * inv);
  }
}

// ---------------------------------------------------------------------------
extern "C" void kernel_launch(void* const* d_in, const int* in_sizes, int n_in,
                              void* d_out, int out_size, void* d_ws, size_t ws_size,
                              hipStream_t stream) {
  (void)in_sizes; (void)n_in; (void)out_size; (void)ws_size;
  const float* query = (const float*)d_in[0];
  const float* key   = (const float*)d_in[1];
  const float* value = (const float*)d_in[2];
  const unsigned char* mask = (const unsigned char*)d_in[3];
  const float* W_A = (const float*)d_in[4];
  const float* W_B = (const float*)d_in[5];
  const float* W_C = (const float*)d_in[6];
  const float* W_D = (const float*)d_in[7];
  const float* W_E = (const float*)d_in[8];
  const float* W_F = (const float*)d_in[9];
  const float* Wq  = (const float*)d_in[10];
  const float* bq  = (const float*)d_in[11];
  const float* Wk  = (const float*)d_in[12];
  const float* bk  = (const float*)d_in[13];
  const float* Wv  = (const float*)d_in[14];
  const float* bv  = (const float*)d_in[15];
  const float* Wo  = (const float*)d_in[16];
  const float* bo  = (const float*)d_in[17];
  float* out = (float*)d_out;

  // workspace layout (bf16 buffers), total 120 MB
  char* ws = (char*)d_ws;
  unsigned short* wWq = (unsigned short*)(ws + (0l   << 20));
  unsigned short* wWk = (unsigned short*)(ws + (2l   << 20));
  unsigned short* wWv = (unsigned short*)(ws + (4l   << 20));
  unsigned short* wWo = (unsigned short*)(ws + (6l   << 20));
  unsigned short* Qb  = (unsigned short*)(ws + (8l   << 20));
  unsigned short* Kt  = (unsigned short*)(ws + (24l  << 20));
  unsigned short* Vb  = (unsigned short*)(ws + (40l  << 20));
  unsigned short* Xb  = (unsigned short*)(ws + (56l  << 20));
  unsigned short* Xq  = (unsigned short*)(ws + (72l  << 20));
  unsigned short* Xk  = (unsigned short*)(ws + (88l  << 20));
  unsigned short* Xv  = (unsigned short*)(ws + (104l << 20));

  const int rows = B_ * S_;  // 8192
  dim3 blk(256);

  // 1) activations -> bf16
  dim3 cgrid((rows * D_) / (256 * 4));
  f32_to_bf16_kernel<<<cgrid, blk, 0, stream>>>(query, Xq);
  f32_to_bf16_kernel<<<cgrid, blk, 0, stream>>>(key,   Xk);
  f32_to_bf16_kernel<<<cgrid, blk, 0, stream>>>(value, Xv);

  // 2) fold LoRA adapters into base weights, convert all weights to bf16
  dim3 wgrid((D_ * HD_) / 256);
  fuse_weights_kernel<<<wgrid, blk, 0, stream>>>(Wq, W_A, W_B, wWq, RNK, HD_);
  fuse_weights_kernel<<<wgrid, blk, 0, stream>>>(Wk, W_C, W_D, wWk, RNK, HD_);
  fuse_weights_kernel<<<wgrid, blk, 0, stream>>>(Wv, W_E, W_F, wWv, RNK, HD_);
  fuse_weights_kernel<<<wgrid, blk, 0, stream>>>(Wo, nullptr, nullptr, wWo, 0, D_);

  // 3) Q/K/V projections (K written pre-transposed for attention)
  dim3 ggrid((rows / 128) * (HD_ / 128));
  gemm_bias_kernel<1><<<ggrid, blk, 0, stream>>>(Xq, wWq, bq, Qb, HD_, D_);
  gemm_bias_kernel<2><<<ggrid, blk, 0, stream>>>(Xk, wWk, bk, Kt, HD_, D_);
  gemm_bias_kernel<1><<<ggrid, blk, 0, stream>>>(Xv, wWv, bv, Vb, HD_, D_);

  // 4) masked flash attention
  attention_kernel<<<dim3(B_ * H_ * (S_ / 128)), blk, 0, stream>>>(Qb, Kt, Vb, mask, Xb);

  // 5) output projection -> f32
  gemm_bias_kernel<0><<<ggrid, blk, 0, stream>>>(Xb, wWo, bo, out, D_, HD_);
}
